// FFN2_42923903156589
// MI455X (gfx1250) — compile-verified, matrix operand order checked by reading, not run on hardware
//
#include <hip/hip_runtime.h>
#include <hip/hip_bf16.h>
#include <math.h>

typedef __attribute__((ext_vector_type(16))) __bf16 v16bf;
typedef __attribute__((ext_vector_type(8)))  __bf16 v8bf;
typedef __attribute__((ext_vector_type(4)))  __bf16 v4bf;
typedef __attribute__((ext_vector_type(8)))  float  v8f;

#define G_     4
#define B_     16
#define CIN_   512
#define T_     2048
#define H_     2048
#define COUT_  512
#define KG1_   128   // Cin/G
#define KG2_   512   // H/G
#define TN_    64    // token tile per block
#define W1ELEMS_ (H_ * KG1_)     // 262144
#define W2ELEMS_ (COUT_ * KG2_)  // 262144

// K8-interleaved LDS layout: element (k, n) -> ((k>>3)*TN_ + n)*8 + (k&7)
__device__ __forceinline__ int k8idx(int k, int n) {
  return ((k >> 3) * TN_ + n) * 8 + (k & 7);
}

// A fragment (16x32 bf16) from row-major bf16 weights: 2x global b128 loads.
__device__ __forceinline__ v16bf load_a_bf16(const __bf16* __restrict__ W,
                                             int row0, int ld, int kb, int lane) {
  const int m = lane & 15, hi = lane >> 4;
  const __bf16* wr = W + (size_t)(row0 + m) * ld + kb + hi * 8;
  const v8bf lo = *(const v8bf*)(wr);
  const v8bf hs = *(const v8bf*)(wr + 16);
  return __builtin_shufflevector(lo, hs, 0, 1, 2, 3, 4, 5, 6, 7,
                                 8, 9, 10, 11, 12, 13, 14, 15);
}

// B fragment (32x16) from K8-interleaved LDS: 2x ds_load_b128.
__device__ __forceinline__ v16bf load_b_k8(const __bf16* S, int kb, int nb, int lane) {
  const int n  = nb + (lane & 15);
  const int hi = lane >> 4;
  const int kg = (kb >> 3) + hi;
  const v8bf lo = *(const v8bf*)(S + ((size_t)kg * TN_ + n) * 8);
  const v8bf hs = *(const v8bf*)(S + ((size_t)(kg + 2) * TN_ + n) * 8);
  return __builtin_shufflevector(lo, hs, 0, 1, 2, 3, 4, 5, 6, 7,
                                 8, 9, 10, 11, 12, 13, 14, 15);
}

// GELU via native CDNA5 v_tanh_f32 transcendental (error << bf16 storage ulp).
__device__ __forceinline__ float gelu_fast(float v) {
  const float v2 = v * v;
  const float u  = v * fmaf(v2, 0.0356774081f, 0.7978845608f);
  float t;
  asm("v_tanh_f32 %0, %1\n\tv_nop\n\tv_nop" : "=v"(t) : "v"(u));
  const float hv = 0.5f * v;
  return fmaf(hv, t, hv);
}

// ---- prologue: convert f32 weights to bf16 in workspace (runs once per call) ----
__global__ __launch_bounds__(256) void cvt_weights(const float* __restrict__ w1,
                                                   const float* __restrict__ w2,
                                                   __bf16* __restrict__ wb) {
  const int idx = blockIdx.x * 256 + threadIdx.x;   // 131072 threads
  const int e   = idx * 4;                          // 524288 elems total
  float4 v;
  if (e < W1ELEMS_) v = *(const float4*)(w1 + e);
  else              v = *(const float4*)(w2 + (e - W1ELEMS_));
  v4bf p;
  p[0] = (__bf16)v.x; p[1] = (__bf16)v.y; p[2] = (__bf16)v.z; p[3] = (__bf16)v.w;
  *(v4bf*)(wb + e) = p;
}

__global__ __launch_bounds__(256) void ffn2_fused_wmma(
    const float* __restrict__ x, const float* __restrict__ mask,
    const __bf16* __restrict__ w1b, const float* __restrict__ b1,
    const __bf16* __restrict__ w2b, const float* __restrict__ b2,
    float* __restrict__ out) {
  __shared__ __bf16 Xs[CIN_ * TN_];  // masked input tile, K8 layout (64 KB)
  __shared__ __bf16 Hs[KG2_ * TN_];  // shuffled+GELU hidden tile, K8 layout (64 KB)

  const int tid  = threadIdx.x;
  const int lane = tid & 31;
  const int wave = tid >> 5;
  const int nlo  = lane & 15;
  const int hi   = lane >> 4;

  const int bid   = blockIdx.x;
  const int tTile = bid & 31;            // T_/TN_ = 32 tiles
  const int g2    = (bid >> 5) & 3;      // output group
  const int b     = bid >> 7;            // batch
  const int t0    = tTile * TN_;

  // ---- X tile load: 8 coalesced b32 loads + mask + pack -> ds_store_b128 ----
  {
    const float* xb = x + (size_t)b * CIN_ * T_;
    const float* mb = mask + (size_t)b * T_;
    for (int q = tid; q < (CIN_ / 8) * TN_; q += 256) {
      const int n  = q & (TN_ - 1);
      const int kg = q >> 6;             // 8-channel group
      const float mk = mb[t0 + n];
      const float* xp = xb + (size_t)(kg * 8) * T_ + t0 + n;
      v8bf pk;
#pragma unroll
      for (int e = 0; e < 8; ++e)
        pk[e] = (__bf16)(xp[(size_t)e * T_] * mk);
      *(v8bf*)(Xs + ((size_t)kg * TN_ + n) * 8) = pk;
    }
  }
  __syncthreads();

  // ---- stage 1: 32 strips of 16 h-rows (4 per wave) ----
  for (int ss = 0; ss < 4; ++ss) {
    const int s  = wave + 8 * ss;        // 0..31
    const int g  = s >> 3;               // input group
    const int ib = (s & 7) * 16;
    const int w1row0 = g * 512 + 128 * g2 + ib;

    // bias folded into accumulator init
    v8f binit;
#pragma unroll
    for (int r = 0; r < 8; ++r) binit[r] = b1[w1row0 + r + hi * 8];
    v8f acc[4] = {binit, binit, binit, binit};

#pragma unroll
    for (int kb = 0; kb < KG1_; kb += 32) {
      const v16bf a = load_a_bf16(w1b, w1row0, KG1_, kb, lane);
#pragma unroll
      for (int nt = 0; nt < 4; ++nt) {
        const v16bf bb = load_b_k8(Xs, g * KG1_ + kb, nt * 16, lane);
        acc[nt] = __builtin_amdgcn_wmma_f32_16x16x32_bf16(
            false, a, false, bb, (short)0, acc[nt], false, false);
      }
    }
    // GELU + channel-shuffled scatter into Hs (K8 layout)
#pragma unroll
    for (int nt = 0; nt < 4; ++nt) {
      const int n = nt * 16 + nlo;
#pragma unroll
      for (int r = 0; r < 8; ++r) {
        const int m = r + hi * 8;
        const float v = gelu_fast(acc[nt][r]);
        const int j = 4 * (ib + m) + g;            // shuffled row in [0,512)
        Hs[k8idx(j, n)] = (__bf16)v;
      }
    }
  }
  __syncthreads();

  // ---- stage 2: out tile 128 x 64, K = 512 over shuffled Hs ----
  const int w2row0 = g2 * 128 + wave * 16;
  v8f binit2;
#pragma unroll
  for (int r = 0; r < 8; ++r) binit2[r] = b2[w2row0 + r + hi * 8];
  v8f acc2[4] = {binit2, binit2, binit2, binit2};

#pragma unroll 4
  for (int kb = 0; kb < KG2_; kb += 32) {
    const v16bf a = load_a_bf16(w2b, w2row0, KG2_, kb, lane);
#pragma unroll
    for (int nt = 0; nt < 4; ++nt) {
      const v16bf bb = load_b_k8(Hs, kb, nt * 16, lane);
      acc2[nt] = __builtin_amdgcn_wmma_f32_16x16x32_bf16(
          false, a, false, bb, (short)0, acc2[nt], false, false);
    }
  }
  const float* mb = mask + (size_t)b * T_;
#pragma unroll
  for (int nt = 0; nt < 4; ++nt) {
    const int n = nt * 16 + nlo;
    const int t = t0 + n;
    const float mk = mb[t];
#pragma unroll
    for (int r = 0; r < 8; ++r) {
      const int oc = w2row0 + r + hi * 8;
      out[((size_t)b * COUT_ + oc) * T_ + t] = acc2[nt][r] * mk;
    }
  }
}

extern "C" void kernel_launch(void* const* d_in, const int* in_sizes, int n_in,
                              void* d_out, int out_size, void* d_ws, size_t ws_size,
                              hipStream_t stream) {
  const float* x    = (const float*)d_in[0];
  const float* mask = (const float*)d_in[1];
  const float* w1   = (const float*)d_in[2];
  const float* b1   = (const float*)d_in[3];
  const float* w2   = (const float*)d_in[4];
  const float* b2   = (const float*)d_in[5];
  float* out = (float*)d_out;

  __bf16* wb  = (__bf16*)d_ws;                  // 1 MB: w1 then w2 as bf16
  const __bf16* w1b = wb;
  const __bf16* w2b = wb + W1ELEMS_;

  cvt_weights<<<(W1ELEMS_ + W2ELEMS_) / (256 * 4), 256, 0, stream>>>(w1, w2, wb);

  const int nBlocks = B_ * G_ * (T_ / TN_);     // 2048
  ffn2_fused_wmma<<<nBlocks, 256, 0, stream>>>(x, mask, w1b, b1, w2b, b2,
                                               (float*)d_out);
  (void)out; (void)in_sizes; (void)n_in; (void)out_size; (void)ws_size;
}
